// BinaryConv_45586782880352
// MI455X (gfx1250) — compile-verified, matrix operand order checked by reading, not run on hardware
//
#include <hip/hip_runtime.h>

// ---------------------------------------------------------------------------
// BinaryConnect CNN forward for MI455X (gfx1250, wave32, WMMA).
// All matmuls use V_WMMA_F32_16X16X32_BF16: sign(w) is exact in bf16,
// activations bf16, accumulation fp32.  A-tiles are staged with
// GLOBAL_LOAD_ASYNC_TO_LDS_B128 (ASYNCcnt) into double-buffered LDS;
// fc1 (K=294912, M=64) uses a deterministic 64-way split-K.
// ---------------------------------------------------------------------------

typedef __attribute__((ext_vector_type(16))) __bf16        v16bf;
typedef __attribute__((ext_vector_type(8)))  float         v8f;
typedef __attribute__((ext_vector_type(8)))  unsigned int  v8u;
typedef __attribute__((ext_vector_type(4)))  int           v4i;

#define AS1 __attribute__((address_space(1)))
#define AS3 __attribute__((address_space(3)))

// fp32 -> bf16 round-to-nearest-even
__device__ __forceinline__ unsigned short f2bf(float f) {
  unsigned int u = __builtin_bit_cast(unsigned int, f);
  u += 0x7FFFu + ((u >> 16) & 1u);
  return (unsigned short)(u >> 16);
}
// jnp.sign semantics (sign(0)==0), result as bf16 bits
__device__ __forceinline__ unsigned short signbf(float f) {
  return f > 0.f ? (unsigned short)0x3F80u
                 : (f < 0.f ? (unsigned short)0xBF80u : (unsigned short)0u);
}
__device__ __forceinline__ unsigned int pack2(unsigned short lo, unsigned short hi) {
  return (unsigned int)lo | ((unsigned int)hi << 16);
}
__device__ __forceinline__ unsigned short cvt_bf(float f)          { return f2bf(f); }
__device__ __forceinline__ unsigned short cvt_bf(unsigned short s) { return s; }

// Async copy 16B global -> LDS (per lane).  Builtin takes (v4i AS1*, v4i AS3*,
// imm offset, imm cpol).  LDS operand = low 32 bits of the generic pointer
// (flat LDS aperture truncates to the DS offset).
__device__ __forceinline__ void async_copy_b128(const unsigned short* g,
                                                unsigned short* l) {
  __builtin_amdgcn_global_load_async_to_lds_b128(
      (AS1 v4i*)(unsigned long long)(__SIZE_TYPE__)g,
      (AS3 v4i*)(unsigned int)(__SIZE_TYPE__)l, 0, 0);
}
__device__ __forceinline__ void wait_asynccnt0() {
  asm volatile("s_wait_asynccnt 0x0" ::: "memory");
}

// LDS tile row stride in 16-bit elements: 32 data + 8 pad = 80B rows.
// Rows stay 16B aligned (b128 LDS ops legal) and 20-dword stride spreads the
// 16 rows of a fragment read over distinct banks.
#define TS 40

// Load one 16x32 bf16 WMMA operand fragment from an LDS tile laid out
// [row][k] (row-major, stride TS).  Matches ISA 7.12.2 16-bit A layout:
//   lanes 0-15 : row = lane,    K-halves {0..7, 16..23}
//   lanes 16-31: row = lane-16, K-halves {8..15, 24..31}
__device__ __forceinline__ v16bf load_frag(const unsigned short* tile,
                                           int rowBase, int lane) {
  const int r = rowBase + (lane & 15);
  const int b = (lane >> 4) << 2;  // dword offset 0 or 4 (K-half 0/8)
  const unsigned int* p = (const unsigned int*)(tile + r * TS);
  v8u u;
  u[0] = p[b + 0]; u[1] = p[b + 1]; u[2] = p[b + 2];  u[3] = p[b + 3];
  u[4] = p[b + 8]; u[5] = p[b + 9]; u[6] = p[b + 10]; u[7] = p[b + 11];
  return __builtin_bit_cast(v16bf, u);
}

// ---------------------------------------------------------------------------
// Generic GEMM:  C[M x N] = A[M x Kloop](bf16) * sign(B[N x Kreal](fp32))^T
// Block tile 64x128, 8 waves (2x4), each wave 2x2 WMMA 16x16x32 tiles.
// A staged via async-to-LDS, B binarized through registers; LDS double
// buffered so next-tile traffic overlaps current-tile WMMAs.
// M must be a multiple of 64 (true at every call site).
// SPLIT=false: direct store with fused bias.
// SPLIT=true : store partial sums to slice blockIdx.z (no bias, no atomics).
// ---------------------------------------------------------------------------
template <bool SPLIT>
__global__ __launch_bounds__(256) void gemm_bin(
    const unsigned short* __restrict__ A, long lda,
    const float* __restrict__ B, long ldb,
    const float* __restrict__ bias,
    float* __restrict__ C, int ldc, long sliceStride,
    int N, int Kloop, int Kreal, int Kchunk) {
  __shared__ __align__(16) unsigned short At[2][64 * TS];
  __shared__ __align__(16) unsigned short Bt[2][128 * TS];

  const int t    = threadIdx.x;
  const int lane = t & 31;
  const int wave = t >> 5;
  const int wm   = wave >> 2;  // 0..1 -> 32-row strip
  const int wn   = wave & 3;   // 0..3 -> 32-col strip
  const long row0 = (long)blockIdx.y * 64;
  const int  col0 = blockIdx.x * 128;
  const int  k0   = blockIdx.z * Kchunk;
  int k1 = k0 + Kchunk; if (k1 > Kloop) k1 = Kloop;

  // global->LDS load assignments
  const int arow = t >> 2, acol = (t & 3) << 3;  // A: 64 rows x 32 k, 8 bf16/thr
  const int brow = t >> 1, bcol = (t & 1) << 4;  // B: 128 rows x 32 k, 16 f32/thr
  const long bn = (long)col0 + brow;

  // Uniform fast-path predicates (per-lane checks only on the cold path).
  const bool fullN      = (col0 + 128 <= N);
  const bool ldbAligned = ((ldb & 3) == 0);

  auto issueA = [&](int kt, int buf) {
    async_copy_b128(A + (row0 + arow) * lda + kt + acol,
                    &At[buf][arow * TS + acol]);
  };
  auto stageB = [&](int kt, unsigned int bw[8]) {
    if (ldbAligned && fullN && (kt + 32 <= Kreal)) {
      const float4* bp = (const float4*)(B + bn * ldb + kt + bcol);
      if (kt + 64 <= Kreal)  // prefetch next K tile -> global_prefetch_b8
        __builtin_prefetch(B + bn * ldb + (kt + 32) + bcol, 0, 1);
#pragma unroll
      for (int q = 0; q < 4; ++q) {
        float4 f = bp[q];
        bw[2 * q + 0] = pack2(signbf(f.x), signbf(f.y));
        bw[2 * q + 1] = pack2(signbf(f.z), signbf(f.w));
      }
    } else {
      unsigned short bs[16];
#pragma unroll
      for (int j = 0; j < 16; ++j) {
        int k = kt + bcol + j;
        float f = (bn < N && k < Kreal) ? B[bn * ldb + k] : 0.f;
        bs[j] = signbf(f);
      }
#pragma unroll
      for (int q = 0; q < 8; ++q) bw[q] = pack2(bs[2 * q], bs[2 * q + 1]);
    }
  };
  auto writeB = [&](int buf, const unsigned int bw[8]) {
    *(uint4*)&Bt[buf][brow * TS + bcol]     = make_uint4(bw[0], bw[1], bw[2], bw[3]);
    *(uint4*)&Bt[buf][brow * TS + bcol + 8] = make_uint4(bw[4], bw[5], bw[6], bw[7]);
  };

  v8f acc[2][2] = {};
  unsigned int bw[8];

  // ---- prologue: stage first tile into buffer 0 ----
  issueA(k0, 0);
  stageB(k0, bw);
  writeB(0, bw);
  wait_asynccnt0();
  __syncthreads();

  int cur = 0;
  for (int kt = k0; kt < k1; kt += 32) {
    const int  nxt     = cur ^ 1;
    const bool hasNext = (kt + 32 < k1);

    if (hasNext) {              // overlap next-tile traffic with WMMAs
      issueA(kt + 32, nxt);     // async -> other LDS buffer
      stageB(kt + 32, bw);      // B global loads + binarize (regs)
    }

    v16bf a0 = load_frag(At[cur], wm * 32 + 0,  lane);
    v16bf a1 = load_frag(At[cur], wm * 32 + 16, lane);
    v16bf b0 = load_frag(Bt[cur], wn * 32 + 0,  lane);
    v16bf b1 = load_frag(Bt[cur], wn * 32 + 16, lane);
    acc[0][0] = __builtin_amdgcn_wmma_f32_16x16x32_bf16(false, a0, false, b0, (short)0, acc[0][0], false, false);
    acc[0][1] = __builtin_amdgcn_wmma_f32_16x16x32_bf16(false, a0, false, b1, (short)0, acc[0][1], false, false);
    acc[1][0] = __builtin_amdgcn_wmma_f32_16x16x32_bf16(false, a1, false, b0, (short)0, acc[1][0], false, false);
    acc[1][1] = __builtin_amdgcn_wmma_f32_16x16x32_bf16(false, a1, false, b1, (short)0, acc[1][1], false, false);

    if (hasNext) {
      writeB(nxt, bw);          // ds writes into the other buffer
      wait_asynccnt0();         // next A tile resident in LDS
    }
    __syncthreads();
    cur = nxt;
  }

  // ---- store per ISA f32 C/D layout: N = lane&15, M = vgpr + 8*(lane>=16) ----
  float* Cs = C + (SPLIT ? (long)blockIdx.z * sliceStride : 0);
  const int nlo = lane & 15;
  const int mhi = (lane >> 4) << 3;
#pragma unroll
  for (int sm = 0; sm < 2; ++sm)
#pragma unroll
    for (int sn = 0; sn < 2; ++sn) {
      int cn = col0 + wn * 32 + sn * 16 + nlo;
      if (!fullN && cn >= N) continue;
      float badd = SPLIT ? 0.f : bias[cn];
#pragma unroll
      for (int v = 0; v < 8; ++v) {
        long cr = row0 + wm * 32 + sm * 16 + mhi + v;
        Cs[cr * ldc + cn] = acc[sm][sn][v] + badd;
      }
    }
}

// ---------------------------------------------------------------------------
// im2col: NCHW (fp32 or bf16) -> A[row = (n,ho,wo)][k = (c,kh,kw)] bf16,
// zero-padded to Kpad (multiple of 32).
// ---------------------------------------------------------------------------
template <typename TIN>
__global__ void im2col_k(const TIN* __restrict__ in, long total, int Kpad,
                         int Kreal, int Cc, int Hin, int Win, int Hout, int Wout,
                         unsigned short* __restrict__ out) {
  long idx = (long)blockIdx.x * blockDim.x + threadIdx.x;
  if (idx >= total) return;
  long row = idx / Kpad;
  int  k   = (int)(idx - row * Kpad);
  unsigned short r16 = 0;
  if (k < Kreal) {
    int c = k / 9, r9 = k - c * 9, kh = r9 / 3, kw = r9 - kh * 3;
    int HWo = Hout * Wout;
    long n  = row / HWo;
    int hw  = (int)(row - n * HWo);
    int ho  = hw / Wout, wo = hw - ho * Wout;
    r16 = cvt_bf(in[((n * Cc + c) * Hin + ho + kh) * (long)Win + wo + kw]);
  }
  out[idx] = r16;
}

// ---------------------------------------------------------------------------
// BatchNorm training-mode stats: per-channel biased mean/var over M rows of
// the [M x N] GEMM output (rows = (n,h,w) exactly the (0,2,3) reduction).
// ---------------------------------------------------------------------------
__global__ __launch_bounds__(256) void bn_stats(const float* __restrict__ C,
                                                long M, int N,
                                                float* __restrict__ mean,
                                                float* __restrict__ rstd) {
  const int c = blockIdx.x, t = threadIdx.x;
  float s = 0.f, ss = 0.f;
  for (long r = t; r < M; r += 256) {
    float v = C[r * N + c];
    s += v; ss += v * v;
  }
  __shared__ float sb[256], qb[256];
  sb[t] = s; qb[t] = ss;
  __syncthreads();
  for (int o = 128; o > 0; o >>= 1) {
    if (t < o) { sb[t] += sb[t + o]; qb[t] += qb[t + o]; }
    __syncthreads();
  }
  if (t == 0) {
    float m   = sb[0] / (float)M;
    float var = qb[0] / (float)M - m * m;
    mean[c] = m;
    rstd[c] = rsqrtf(var + 1e-5f);
  }
}

// normalize + affine + ReLU, write bf16 NCHW (next layer's input / fc1 input)
__global__ void bn_norm_to_nchw(const float* __restrict__ C, long total, int Cc,
                                int HW, int ldc, const float* __restrict__ mean,
                                const float* __restrict__ rstd,
                                const float* __restrict__ gamma,
                                const float* __restrict__ beta,
                                unsigned short* __restrict__ out) {
  long idx = (long)blockIdx.x * blockDim.x + threadIdx.x;
  if (idx >= total) return;
  int chw = Cc * HW;
  long n  = idx / chw;
  int rem = (int)(idx - n * chw);
  int c   = rem / HW;
  int hw  = rem - c * HW;
  long row = n * HW + hw;
  float v = C[row * ldc + c];
  v = (v - mean[c]) * rstd[c] * gamma[c] + beta[c];
  v = fmaxf(v, 0.f);
  out[idx] = f2bf(v);
}

// deterministic split-K reduction + bias + ReLU epilogues
__global__ void fc_reduce_bf16(const float* __restrict__ P, int nz, int total,
                               int Nfc, const float* __restrict__ bias,
                               unsigned short* __restrict__ out) {
  int idx = blockIdx.x * blockDim.x + threadIdx.x;
  if (idx >= total) return;
  float s = 0.f;
  for (int z = 0; z < nz; ++z) s += P[(long)z * total + idx];
  s = fmaxf(s + bias[idx % Nfc], 0.f);
  out[idx] = f2bf(s);
}
__global__ void fc_reduce_f32(const float* __restrict__ P, int nz, int total,
                              int Nfc, const float* __restrict__ bias,
                              float* __restrict__ out) {
  int idx = blockIdx.x * blockDim.x + threadIdx.x;
  if (idx >= total) return;
  float s = 0.f;
  for (int z = 0; z < nz; ++z) s += P[(long)z * total + idx];
  out[idx] = fmaxf(s + bias[idx % Nfc], 0.f);
}

// ---------------------------------------------------------------------------
extern "C" void kernel_launch(void* const* d_in, const int* in_sizes, int n_in,
                              void* d_out, int out_size, void* d_ws,
                              size_t ws_size, hipStream_t stream) {
  (void)in_sizes; (void)n_in; (void)out_size; (void)ws_size;

  const float* x     = (const float*)d_in[0];
  const float* w[4]  = {(const float*)d_in[1], (const float*)d_in[5],
                        (const float*)d_in[9], (const float*)d_in[13]};
  const float* cb[4] = {(const float*)d_in[2], (const float*)d_in[6],
                        (const float*)d_in[10], (const float*)d_in[14]};
  const float* gm[4] = {(const float*)d_in[3], (const float*)d_in[7],
                        (const float*)d_in[11], (const float*)d_in[15]};
  const float* bt[4] = {(const float*)d_in[4], (const float*)d_in[8],
                        (const float*)d_in[12], (const float*)d_in[16]};
  const float* fw1 = (const float*)d_in[17]; const float* fb1 = (const float*)d_in[18];
  const float* fw2 = (const float*)d_in[19]; const float* fb2 = (const float*)d_in[20];
  const float* fw3 = (const float*)d_in[21]; const float* fb3 = (const float*)d_in[22];

  // ---- scratch carve-out (~300 MB) ----
  char* ws = (char*)d_ws;
  size_t off = 0;
  auto carve = [&](size_t bytes) -> void* {
    void* p = ws + off;
    off += (bytes + 255) & ~(size_t)255;
    return p;
  };
  unsigned short* Abuf = (unsigned short*)carve(36864UL * 2304 * 2);       // im2col (max: conv4)
  float*          Cbuf = (float*)carve(36864UL * 512 * 4);                 // conv GEMM out
  unsigned short* act  = (unsigned short*)carve(64UL * 512 * 24 * 24 * 2); // bf16 activations
  float*          P1   = (float*)carve(64UL * 64 * 1024 * 4);              // fc split-K partials
  unsigned short* fcA  = (unsigned short*)carve(64UL * 1024 * 2);
  unsigned short* fcB  = (unsigned short*)carve(64UL * 1024 * 2);
  float*          P3   = (float*)carve(64UL * 10 * 4);
  float*          meanb = (float*)carve(512 * 4);
  float*          rstdb = (float*)carve(512 * 4);

  // ---- conv stack ----
  const int Cin_[4]  = {3, 128, 128, 256};
  const int Hin_[4]  = {32, 30, 28, 26};
  const int Cout_[4] = {128, 128, 256, 512};

  for (int l = 0; l < 4; ++l) {
    const int Cin = Cin_[l], Hin = Hin_[l], Win = Hin;
    const int Cout = Cout_[l], Hout = Hin - 2, Wout = Hout;
    const int Kreal = Cin * 9;
    const int Kpad  = (Kreal + 31) & ~31;
    const long M    = 64L * Hout * Wout;  // multiple of 64 for all layers
    const long tot  = M * Kpad;
    const unsigned gIm = (unsigned)((tot + 255) / 256);

    if (l == 0)
      im2col_k<float><<<gIm, 256, 0, stream>>>(x, tot, Kpad, Kreal, Cin, Hin,
                                               Win, Hout, Wout, Abuf);
    else
      im2col_k<unsigned short><<<gIm, 256, 0, stream>>>(act, tot, Kpad, Kreal,
                                                        Cin, Hin, Win, Hout,
                                                        Wout, Abuf);

    dim3 gG(Cout / 128, (unsigned)(M / 64), 1);
    gemm_bin<false><<<gG, 256, 0, stream>>>(Abuf, Kpad, w[l], Kreal, cb[l],
                                            Cbuf, Cout, 0, Cout, Kpad, Kreal,
                                            Kpad);

    bn_stats<<<Cout, 256, 0, stream>>>(Cbuf, M, Cout, meanb, rstdb);

    const long atot = M * Cout;
    bn_norm_to_nchw<<<(unsigned)((atot + 255) / 256), 256, 0, stream>>>(
        Cbuf, atot, Cout, Hout * Wout, Cout, meanb, rstdb, gm[l], bt[l], act);
  }

  // ---- fc1: [64 x 294912] x sign(fw1)^T, 64-way split-K (HBM saturation) ----
  {
    const int ZK = 64, Kc = 294912 / ZK;  // 4608 = 144 K-steps per slice
    dim3 g1(1024 / 128, 1, ZK);
    gemm_bin<true><<<g1, 256, 0, stream>>>(act, 294912, fw1, 294912, nullptr,
                                           P1, 1024, 64L * 1024, 1024, 294912,
                                           294912, Kc);
    fc_reduce_bf16<<<(64 * 1024 + 255) / 256, 256, 0, stream>>>(
        P1, ZK, 64 * 1024, 1024, fb1, fcA);
  }
  // ---- fc2 ----
  {
    dim3 g2(1024 / 128, 1, 1);
    gemm_bin<true><<<g2, 256, 0, stream>>>(fcA, 1024, fw2, 1024, nullptr, P1,
                                           1024, 64L * 1024, 1024, 1024, 1024,
                                           1024);
    fc_reduce_bf16<<<(64 * 1024 + 255) / 256, 256, 0, stream>>>(
        P1, 1, 64 * 1024, 1024, fb2, fcB);
  }
  // ---- fc3 (N=10, handled by bounds checks) ----
  {
    dim3 g3(1, 1, 1);
    gemm_bin<true><<<g3, 256, 0, stream>>>(fcB, 1024, fw3, 1024, nullptr, P3,
                                           10, 64L * 10, 10, 1024, 1024, 1024);
    fc_reduce_f32<<<(640 + 255) / 256, 256, 0, stream>>>(P3, 1, 640, 10, fb3,
                                                         (float*)d_out);
  }
}
